// CCNF_16664473108884
// MI455X (gfx1250) — matrix operand order repaired and checked by reading, not compiled, not run on hardware
//
#include <hip/hip_runtime.h>
#include <hip/hip_bf16.h>

typedef __bf16 bf16_t;
typedef __attribute__((ext_vector_type(16))) __bf16 v16bf;
typedef __attribute__((ext_vector_type(8)))  float  v8f;
typedef __attribute__((ext_vector_type(4)))  int    v4i;

#define B_ROWS 2048
#define D_DIM  32
#define C_DIM  128
#define H_DIM  512
#define H2_DIM 1024
#define IN_DIM 161     // D + 1 + C
#define K1PAD  192     // layer-1 K padded to multiple of 32
#define XPAD   200     // LDS row stride (elements) for x  : 400B, 16B-aligned, bank-friendly
#define HPAD   520     // LDS row stride (elements) for h  : 1040B, 16B-aligned, bank-friendly
#define MT     16      // batch rows per workgroup
#define NTHREADS 256

union FragB16 { v16bf v; v4i q[2]; };

__device__ __forceinline__ float fast_sigmoid_mul(float a, float b) {
    // a * sigmoid(b) using v_exp_f32 + v_rcp_f32 (avoid IEEE div sequence)
    return a * __builtin_amdgcn_rcpf(1.0f + __expf(-b));
}

// ---------------------------------------------------------------------------
// Pre-pass: fp32 [K][N] row-major  ->  bf16 transposed [N][Kpad], zero-padded K
// ---------------------------------------------------------------------------
__global__ void ccnf_convert_transpose(const float* __restrict__ src,
                                       bf16_t* __restrict__ dst,
                                       int K, int N, int Kpad) {
    int idx = blockIdx.x * blockDim.x + threadIdx.x;
    if (idx >= N * Kpad) return;
    int n = idx / Kpad;
    int k = idx - n * Kpad;
    float v = (k < K) ? src[k * N + n] : 0.0f;
    dst[idx] = (bf16_t)v;
}

// ---------------------------------------------------------------------------
// Persistent RK4 kernel: one workgroup integrates 16 batch rows through all
// n_steps RK4 steps. All activations live in LDS/VGPRs; weights stream from
// L2-resident bf16 copies via WMMA B-fragments.
// ---------------------------------------------------------------------------
__global__ __launch_bounds__(NTHREADS) void ccnf_rk4_kernel(
    const float* __restrict__ theta0,
    const float* __restrict__ context,
    const float* __restrict__ b1,
    const float* __restrict__ b2,
    const float* __restrict__ b3,
    const bf16_t* __restrict__ W1T,   // [1024][K1PAD]
    const bf16_t* __restrict__ W2T,   // [1024][512]
    const bf16_t* __restrict__ W3T,   // [32][512]
    const int* __restrict__ nsteps_p,
    float* __restrict__ out)
{
    __shared__ bf16_t xs [MT * XPAD];    // network input rows (bf16)
    __shared__ bf16_t h1s[MT * HPAD];    // hidden 1 (bf16)
    __shared__ bf16_t h2s[MT * HPAD];    // hidden 2 (bf16)
    __shared__ float  th  [MT * D_DIM];  // current theta
    __shared__ float  stg [MT * D_DIM];  // stage input theta
    __shared__ float  kacc[MT * D_DIM];  // RK4 k accumulator
    __shared__ float  vfo [MT * D_DIM];  // vector-field output

    const int tid  = threadIdx.x;
    const int lane = tid & 31;
    const int wid  = tid >> 5;          // wave id 0..7
    const int m16  = lane & 15;
    const int hfl  = lane >> 4;         // which K-half this lane holds
    const int row0 = blockIdx.x * MT;

    // ---- one-time init: theta, context columns, zero K-padding ----
    for (int i = tid; i < MT * D_DIM; i += NTHREADS)
        th[i] = theta0[row0 * D_DIM + i];
    for (int i = tid; i < MT * XPAD; i += NTHREADS) {
        int m = i / XPAD, c = i - m * XPAD;
        if (c >= 33 && c < 33 + C_DIM)
            xs[i] = (bf16_t)context[(row0 + m) * C_DIM + (c - 33)];
        else if (c >= IN_DIM)
            xs[i] = (bf16_t)0.0f;
    }

    // ---- hoist biases: invariant across all 64 evaluations ----
    float b1A[4], b1B[4], b2A[4], b2B[4];
    #pragma unroll
    for (int j = 0; j < 4; ++j) {
        const int ntA = wid * 4 + j;
        b1A[j] = b1[ntA * 16 + m16];
        b1B[j] = b1[(32 + ntA) * 16 + m16];
        b2A[j] = b2[ntA * 16 + m16];
        b2B[j] = b2[(32 + ntA) * 16 + m16];
    }
    const float b3v = (wid < 2) ? b3[wid * 16 + m16] : 0.0f;

    const int   nsteps = nsteps_p[0];
    const float dt     = 1.0f / (float)nsteps;

    // -------- one vector-field evaluation: stg -> vfo --------
    auto vf_eval = [&](float t) {
        __syncthreads();
        // theta + time columns of x
        for (int i = tid; i < MT * 33; i += NTHREADS) {
            int m = i / 33, c = i - m * 33;
            xs[m * XPAD + c] = (c < D_DIM) ? (bf16_t)stg[m * D_DIM + c] : (bf16_t)t;
        }
        __syncthreads();

        // ===== layer 1: GLU(x @ W1 + b1) -> h1s =====
        {
            v8f zero = {};
            v8f acc[8];
            #pragma unroll
            for (int j = 0; j < 8; ++j) acc[j] = zero;

            #pragma unroll 2
            for (int ks = 0; ks < K1PAD / 32; ++ks) {
                FragB16 a;
                const bf16_t* ap = &xs[m16 * XPAD + ks * 32 + 8 * hfl];
                a.q[0] = *(const v4i*)ap;
                a.q[1] = *(const v4i*)(ap + 16);
                #pragma unroll
                for (int j = 0; j < 8; ++j) {
                    const int nt = (j < 4) ? (wid * 4 + j) : (32 + wid * 4 + (j - 4));
                    const bf16_t* bp = W1T + (nt * 16 + m16) * K1PAD + ks * 32 + 8 * hfl;
                    FragB16 b;
                    b.q[0] = *(const v4i*)bp;
                    b.q[1] = *(const v4i*)(bp + 16);
                    acc[j] = __builtin_amdgcn_wmma_f32_16x16x32_bf16(
                        false, a.v, false, b.v, (short)0, acc[j], false, false);
                }
            }
            #pragma unroll
            for (int j = 0; j < 4; ++j) {
                const int ntA = wid * 4 + j;
                #pragma unroll
                for (int r = 0; r < 8; ++r) {
                    float hv = fast_sigmoid_mul(acc[j][r] + b1A[j], acc[j + 4][r] + b1B[j]);
                    int mrow = r + 8 * hfl;
                    h1s[mrow * HPAD + ntA * 16 + m16] = (bf16_t)hv;
                }
            }
        }
        __syncthreads();

        // ===== layer 2: GLU(h1 @ W2 + b2) -> h2s =====
        {
            v8f zero = {};
            v8f acc[8];
            #pragma unroll
            for (int j = 0; j < 8; ++j) acc[j] = zero;

            #pragma unroll 2
            for (int ks = 0; ks < H_DIM / 32; ++ks) {
                FragB16 a;
                const bf16_t* ap = &h1s[m16 * HPAD + ks * 32 + 8 * hfl];
                a.q[0] = *(const v4i*)ap;
                a.q[1] = *(const v4i*)(ap + 16);
                #pragma unroll
                for (int j = 0; j < 8; ++j) {
                    const int nt = (j < 4) ? (wid * 4 + j) : (32 + wid * 4 + (j - 4));
                    const bf16_t* bp = W2T + (nt * 16 + m16) * H_DIM + ks * 32 + 8 * hfl;
                    FragB16 b;
                    b.q[0] = *(const v4i*)bp;
                    b.q[1] = *(const v4i*)(bp + 16);
                    acc[j] = __builtin_amdgcn_wmma_f32_16x16x32_bf16(
                        false, a.v, false, b.v, (short)0, acc[j], false, false);
                }
            }
            #pragma unroll
            for (int j = 0; j < 4; ++j) {
                const int ntA = wid * 4 + j;
                #pragma unroll
                for (int r = 0; r < 8; ++r) {
                    float hv = fast_sigmoid_mul(acc[j][r] + b2A[j], acc[j + 4][r] + b2B[j]);
                    int mrow = r + 8 * hfl;
                    h2s[mrow * HPAD + ntA * 16 + m16] = (bf16_t)hv;
                }
            }
        }
        __syncthreads();

        // ===== layer 3: h2 @ W3 + b3 -> vfo (only 2 N-tiles; waves 0,1) =====
        if (wid < 2) {
            v8f acc3 = {};
            #pragma unroll 2
            for (int ks = 0; ks < H_DIM / 32; ++ks) {
                FragB16 a;
                const bf16_t* ap = &h2s[m16 * HPAD + ks * 32 + 8 * hfl];
                a.q[0] = *(const v4i*)ap;
                a.q[1] = *(const v4i*)(ap + 16);
                const bf16_t* bp = W3T + (wid * 16 + m16) * H_DIM + ks * 32 + 8 * hfl;
                FragB16 b;
                b.q[0] = *(const v4i*)bp;
                b.q[1] = *(const v4i*)(bp + 16);
                acc3 = __builtin_amdgcn_wmma_f32_16x16x32_bf16(
                    false, a.v, false, b.v, (short)0, acc3, false, false);
            }
            #pragma unroll
            for (int r = 0; r < 8; ++r) {
                int mrow = r + 8 * hfl;
                vfo[mrow * D_DIM + wid * 16 + m16] = acc3[r] + b3v;
            }
        }
        __syncthreads();
    };

    // -------- RK4 time loop --------
    for (int s = 0; s < nsteps; ++s) {
        const float t0 = (float)s * dt;

        for (int i = tid; i < MT * D_DIM; i += NTHREADS) stg[i] = th[i];
        vf_eval(t0);                                   // k1
        for (int i = tid; i < MT * D_DIM; i += NTHREADS) {
            float k = vfo[i]; kacc[i] = k; stg[i] = th[i] + 0.5f * dt * k;
        }
        vf_eval(t0 + 0.5f * dt);                       // k2
        for (int i = tid; i < MT * D_DIM; i += NTHREADS) {
            float k = vfo[i]; kacc[i] += 2.0f * k; stg[i] = th[i] + 0.5f * dt * k;
        }
        vf_eval(t0 + 0.5f * dt);                       // k3
        for (int i = tid; i < MT * D_DIM; i += NTHREADS) {
            float k = vfo[i]; kacc[i] += 2.0f * k; stg[i] = th[i] + dt * k;
        }
        vf_eval(t0 + dt);                              // k4
        for (int i = tid; i < MT * D_DIM; i += NTHREADS) {
            th[i] += (dt / 6.0f) * (kacc[i] + vfo[i]);
        }
    }

    __syncthreads();
    for (int i = tid; i < MT * D_DIM; i += NTHREADS)
        out[row0 * D_DIM + i] = th[i];
}

// ---------------------------------------------------------------------------
extern "C" void kernel_launch(void* const* d_in, const int* in_sizes, int n_in,
                              void* d_out, int out_size, void* d_ws, size_t ws_size,
                              hipStream_t stream) {
    const float* theta0  = (const float*)d_in[0];
    const float* context = (const float*)d_in[1];
    const float* W1      = (const float*)d_in[2];
    const float* b1      = (const float*)d_in[3];
    const float* W2      = (const float*)d_in[4];
    const float* b2      = (const float*)d_in[5];
    const float* W3      = (const float*)d_in[6];
    const float* b3      = (const float*)d_in[7];
    const int*   nsteps  = (const int*)d_in[8];

    char* ws = (char*)d_ws;
    bf16_t* W1T = (bf16_t*)(ws);
    bf16_t* W2T = (bf16_t*)(ws + (size_t)H2_DIM * K1PAD * sizeof(bf16_t));
    bf16_t* W3T = (bf16_t*)(ws + (size_t)H2_DIM * K1PAD * sizeof(bf16_t)
                               + (size_t)H2_DIM * H_DIM * sizeof(bf16_t));

    const int n1 = H2_DIM * K1PAD;
    const int n2 = H2_DIM * H_DIM;
    const int n3 = D_DIM * H_DIM;
    ccnf_convert_transpose<<<(n1 + 255) / 256, 256, 0, stream>>>(W1, W1T, IN_DIM, H2_DIM, K1PAD);
    ccnf_convert_transpose<<<(n2 + 255) / 256, 256, 0, stream>>>(W2, W2T, H_DIM, H2_DIM, H_DIM);
    ccnf_convert_transpose<<<(n3 + 255) / 256, 256, 0, stream>>>(W3, W3T, H_DIM, D_DIM, H_DIM);

    ccnf_rk4_kernel<<<B_ROWS / MT, NTHREADS, 0, stream>>>(
        theta0, context, b1, b2, b3, W1T, W2T, W3T, nsteps, (float*)d_out);
}
